// bPC_CNN_83674552860712
// MI455X (gfx1250) — matrix operand order repaired and checked by reading, not compiled, not run on hardware
//
#include <hip/hip_runtime.h>

// ---------------------------------------------------------------------------
// bPC CNN (predictive coding) on MI455X / gfx1250.
// Every conv / transposed-conv / FC (forward + adjoint) is an implicit GEMM
//   C[M,N] = A[M,K] * B[K,N]
// through v_wmma_f32_16x16x32_bf16.  A is gathered on the fly into LDS (bf16),
// B is a pre-transposed bf16 weight panel [Npad][Kpad] (K contiguous).
// The GEMM is templated on the A-loader MODE so each instantiation carries a
// single branch-free gather path.  Double-buffered LDS + register staging:
// global loads for tile t+1 issue before the WMMAs of tile t; all fragment
// ds_load_b128s are hoisted ahead of the 4 back-to-back WMMAs.
// ---------------------------------------------------------------------------

typedef __attribute__((ext_vector_type(16))) __bf16 v16bf;
typedef __attribute__((ext_vector_type(8)))  __bf16 v8bf;
typedef __attribute__((ext_vector_type(4)))  __bf16 v4bf;
typedef __attribute__((ext_vector_type(8)))  float  v8f;
typedef __attribute__((ext_vector_type(4)))  float  v4f;

#define AD   0.2f
#define AG   0.8f
#define LRc  0.1f
#define MOMc 0.9f
#define NSTEPS 20

// MODE: 0 dense bf16 | 1 dense f32+relu | 2 conv Cp>=32 f32+relu
//       3 conv Cp>=32 bf16 | 4 conv Cp==4 f32+relu | 5 conv Cp==4 bf16
struct ConvDesc {
  const void* src;
  int Cin, lcp;      // input channels (addressing), log2(padded channels)
  int H, W;          // input spatial
  int low, lohw;     // log2(OW), log2(OH*OW)
  int s, ldil, pl;   // stride, log2(lhs dilation) (0|1), low pad
};

#define MB 128
#define NB 32
#define KB 64
#define AS 72
#define BS 72

template<bool RELU>
__device__ __forceinline__ v8bf cvt8(v4f a, v4f b) {
  v8bf r;
  #pragma unroll
  for (int j = 0; j < 4; ++j) {
    float x = a[j], y = b[j];
    if (RELU) { x = fmaxf(x, 0.f); y = fmaxf(y, 0.f); }
    r[j] = (__bf16)x; r[j + 4] = (__bf16)y;
  }
  return r;
}

struct StageRegs { v8bf bB; v8bf ra[4]; };

template<int MODE>
__global__ __launch_bounds__(256)
void gemm_wmma(ConvDesc ad, const __bf16* __restrict__ Bw,
               int M, int Kpad,
               const float* __restrict__ bias,
               const float* __restrict__ subf,   // out = subf - (A*B+bias)
               int N, int ldc,
               float* __restrict__ outF, __bf16* __restrict__ outB)
{
  __shared__ __bf16 As_[2][MB * AS];   // 2 x 18 KB
  __shared__ __bf16 Bs_[2][NB * BS];   // 2 x 4.5 KB

  const int tid  = threadIdx.x;
  const int lane = tid & 31;
  const int wid  = tid >> 5;
  const int m0 = blockIdx.x * MB;
  const int n0 = blockIdx.y * NB;

  const int rA = tid >> 1;           // A row owned by this thread
  const int kh = (tid & 1) << 5;     // 32 contiguous k
  const int mg = m0 + rA;
  const int nB  = tid >> 3;          // B row (n) owned
  const int koB = (tid & 7) << 3;    // 8 contiguous k

  int b = 0, oy = 0, ox = 0;
  if (MODE >= 2) {
    b = mg >> ad.lohw;
    int rem = mg & ((1 << ad.lohw) - 1);
    oy = rem >> ad.low;
    ox = rem & ((1 << ad.low) - 1);
  }
  const int Cm = (1 << ad.lcp) - 1;
  const int dm = (1 << ad.ldil) - 1;
  const __bf16 bz = (__bf16)0.f;

  auto loadRegs = [&](int k0) -> StageRegs {
    StageRegs st;
    st.bB = *(const v8bf*)(Bw + (size_t)(n0 + nB) * Kpad + k0 + koB);
    if (k0 + KB < Kpad)
      __builtin_prefetch((const void*)(Bw + (size_t)(n0 + nB) * Kpad + k0 + KB + koB), 0, 3);
    v8bf z = {bz, bz, bz, bz, bz, bz, bz, bz};
    st.ra[0] = z; st.ra[1] = z; st.ra[2] = z; st.ra[3] = z;

    if (MODE == 0) {                       // dense bf16
      if (mg < M) {
        const v8bf* ap = (const v8bf*)((const __bf16*)ad.src + (size_t)mg * Kpad + k0 + kh);
        st.ra[0] = ap[0]; st.ra[1] = ap[1]; st.ra[2] = ap[2]; st.ra[3] = ap[3];
      }
    } else if (MODE == 1) {                // dense f32 + relu
      if (mg < M) {
        const v4f* ap = (const v4f*)((const float*)ad.src + (size_t)mg * Kpad + k0 + kh);
        #pragma unroll
        for (int q = 0; q < 4; ++q) st.ra[q] = cvt8<true>(ap[2 * q], ap[2 * q + 1]);
      }
    } else if (MODE == 2 || MODE == 3) {   // conv, Cp>=32: one tap / 32-chunk
      int kk0 = k0 + kh;
      int p  = kk0 >> ad.lcp;
      int cb = kk0 & Cm;
      int ky = p / 3, kx = p - ky * 3;
      int iyn = oy * ad.s + ky - ad.pl;
      int ixn = ox * ad.s + kx - ad.pl;
      bool ok = (p < 9) && (mg < M) && iyn >= 0 && ixn >= 0 &&
                !(iyn & dm) && !(ixn & dm);
      int iy = iyn >> ad.ldil, ix = ixn >> ad.ldil;
      ok = ok && iy < ad.H && ix < ad.W;
      if (ok) {
        size_t idx = (((size_t)b * ad.H + iy) * ad.W + ix) * ad.Cin + cb;
        if (MODE == 3) {
          const v8bf* sp = (const v8bf*)((const __bf16*)ad.src + idx);
          st.ra[0] = sp[0]; st.ra[1] = sp[1]; st.ra[2] = sp[2]; st.ra[3] = sp[3];
        } else {
          const v4f* sp = (const v4f*)((const float*)ad.src + idx);
          #pragma unroll
          for (int q = 0; q < 4; ++q) st.ra[q] = cvt8<true>(sp[2 * q], sp[2 * q + 1]);
        }
      }
    } else {                               // conv, Cp==4: 8 taps / chunk, 4-ch pixels
      #pragma unroll
      for (int q = 0; q < 8; ++q) {
        int p = (kh >> 2) + q;             // k0 == 0 for these ops (Kpad==64)
        int ky = p / 3, kx = p - ky * 3;
        int iyn = oy * ad.s + ky - ad.pl;
        int ixn = ox * ad.s + kx - ad.pl;
        bool ok = (p < 9) && (mg < M) && iyn >= 0 && ixn >= 0 &&
                  iyn < ad.H && ixn < ad.W;
        int base = (q & 1) * 4;
        if (MODE == 4) {
          v4f f = {0.f, 0.f, 0.f, 0.f};
          if (ok) f = *(const v4f*)((const float*)ad.src +
                        (((size_t)b * ad.H + iyn) * ad.W + ixn) * 4);
          #pragma unroll
          for (int j = 0; j < 4; ++j) st.ra[q >> 1][base + j] = (__bf16)fmaxf(f[j], 0.f);
        } else {
          v4bf f = {bz, bz, bz, bz};
          if (ok) f = *(const v4bf*)((const __bf16*)ad.src +
                        (((size_t)b * ad.H + iyn) * ad.W + ixn) * 4);
          #pragma unroll
          for (int j = 0; j < 4; ++j) st.ra[q >> 1][base + j] = f[j];
        }
      }
    }
    return st;
  };
  auto storeRegs = [&](const StageRegs& st, int bufi) {
    *(v8bf*)(Bs_[bufi] + nB * BS + koB) = st.bB;
    v8bf* dp = (v8bf*)(As_[bufi] + rA * AS + kh);
    dp[0] = st.ra[0]; dp[1] = st.ra[1]; dp[2] = st.ra[2]; dp[3] = st.ra[3];
  };

  v8f acc0 = {0.f,0.f,0.f,0.f,0.f,0.f,0.f,0.f};
  v8f acc1 = {0.f,0.f,0.f,0.f,0.f,0.f,0.f,0.f};
  const int arow  = (wid << 4) + (lane & 15);
  const int ahalf = lane >> 4;
  const int bn    = lane & 15;

  const int nk = Kpad / KB;
  {
    StageRegs st = loadRegs(0);
    storeRegs(st, 0);
  }
  __syncthreads();
  for (int t = 0; t < nk; ++t) {
    StageRegs nxt;
    const bool more = (t + 1) < nk;
    if (more) nxt = loadRegs((t + 1) * KB);

    const __bf16* arp = As_[t & 1] + arow * AS;
    const __bf16* bsp = Bs_[t & 1];

    // hoist all 12 fragment ds_load_b128s, then issue 4 WMMAs back-to-back
    v16bf afr[2], bfr0[2], bfr1[2];
    #pragma unroll
    for (int u = 0; u < 2; ++u) {
      const int ks = u * 32;
      v8bf a0 = *(const v8bf*)(arp + ks + 8 * ahalf);
      v8bf a1 = *(const v8bf*)(arp + ks + 16 + 8 * ahalf);
      afr[u] = __builtin_shufflevector(a0, a1, 0,1,2,3,4,5,6,7,8,9,10,11,12,13,14,15);

      const __bf16* bp0 = bsp + bn * BS + ks + 8 * ahalf;
      v8bf b0 = *(const v8bf*)bp0;
      v8bf b1 = *(const v8bf*)(bp0 + 16);
      bfr0[u] = __builtin_shufflevector(b0, b1, 0,1,2,3,4,5,6,7,8,9,10,11,12,13,14,15);

      const __bf16* bp1 = bsp + (16 + bn) * BS + ks + 8 * ahalf;
      v8bf c0 = *(const v8bf*)bp1;
      v8bf c1 = *(const v8bf*)(bp1 + 16);
      bfr1[u] = __builtin_shufflevector(c0, c1, 0,1,2,3,4,5,6,7,8,9,10,11,12,13,14,15);
    }
    #pragma unroll
    for (int u = 0; u < 2; ++u) {
      acc0 = __builtin_amdgcn_wmma_f32_16x16x32_bf16(false, afr[u], false, bfr0[u], (short)0, acc0, false, false);
      acc1 = __builtin_amdgcn_wmma_f32_16x16x32_bf16(false, afr[u], false, bfr1[u], (short)0, acc1, false, false);
    }
    if (more) storeRegs(nxt, (t + 1) & 1);
    __syncthreads();
  }

  #pragma unroll
  for (int t = 0; t < 2; ++t) {
    v8f acc = t ? acc1 : acc0;
    int ncol = n0 + t * 16 + bn;
    #pragma unroll
    for (int r = 0; r < 8; ++r) {
      int mrow = m0 + (wid << 4) + r + ((lane >> 4) << 3);
      if (mrow < M && ncol < N) {
        float vv = acc[r] + (bias ? bias[ncol] : 0.f);
        if (subf) vv = subf[(size_t)mrow * ldc + ncol] - vv;
        if (outF) outF[(size_t)mrow * ldc + ncol] = vv;
        else      outB[(size_t)mrow * ldc + ncol] = (__bf16)vv;
      }
    }
  }
}

// ---------------------------------------------------------------------------
// elementwise kernels
// ---------------------------------------------------------------------------
__global__ void k_zero(float* p, size_t n) {
  size_t i = (size_t)blockIdx.x * blockDim.x + threadIdx.x;
  if (i < n) p[i] = 0.f;
}
__global__ void k_zerobf(__bf16* p, size_t n) {
  size_t i = (size_t)blockIdx.x * blockDim.x + threadIdx.x;
  if (i < n) p[i] = (__bf16)0.f;
}

__global__ void k_nchw2nhwc(float* dst, const float* src, int B, int C, int H, int W, int Cd) {
  size_t i = (size_t)blockIdx.x * blockDim.x + threadIdx.x;
  size_t tot = (size_t)B * C * H * W;
  if (i >= tot) return;
  int w = i % W; size_t t = i / W;
  int h = t % H; t /= H;
  int c = t % C; int b = (int)(t / C);
  dst[(((size_t)b * H + h) * W + w) * Cd + c] = src[i];
}

__global__ void k_update(float* __restrict__ x, float* __restrict__ v,
                         const __bf16* __restrict__ e, const __bf16* __restrict__ f,
                         const float* __restrict__ adjd, const float* __restrict__ adjg,
                         size_t n)
{
  size_t i = (size_t)blockIdx.x * blockDim.x + threadIdx.x;
  if (i >= n) return;
  float xo = x[i];
  float mask = xo > 0.f ? 1.f : 0.f;
  float g = AD * ((float)e[i] - mask * adjd[i]) +
            AG * ((float)f[i] - mask * adjg[i]);
  float nv = MOMc * v[i] + g;
  v[i] = nv;
  x[i] = xo - LRc * nv;
}

__global__ void k_pack(__bf16* dst, const float* src, int M, int Kreal, int Kpad, int relu) {
  int idx = blockIdx.x * 256 + threadIdx.x;
  if (idx >= M * Kpad) return;
  int m = idx / Kpad, k = idx - m * Kpad;
  float v = (k < Kreal) ? src[(size_t)m * Kreal + k] : 0.f;
  if (relu && v < 0.f) v = 0.f;
  dst[idx] = (__bf16)v;
}

// Weight panels stored TRANSPOSED: dst[n * Kpad + k].
// mode 0: forward conv, w OIHW:           val = w[o,c,ky,kx]
// mode 1: transposed conv, w (Cin,O,3,3): val = w[c,o,2-ky,2-kx]
__global__ void k_prep_w(__bf16* dst, const float* w, int Cin, int lcp, int O,
                         int Npad, int Kpad, int mode)
{
  int idx = blockIdx.x * 256 + threadIdx.x;
  if (idx >= Npad * Kpad) return;
  int n = idx / Kpad, k = idx - n * Kpad;
  int Cp = 1 << lcp;
  int c = k & (Cp - 1), p = k >> lcp;
  float val = 0.f;
  if (p < 9 && c < Cin && n < O) {
    int ky = p / 3, kx = p - ky * 3;
    if (mode == 0) val = w[(((size_t)n * Cin + c) * 3 + ky) * 3 + kx];
    else           val = w[(((size_t)c * O + n) * 3 + (2 - ky)) * 3 + (2 - kx)];
  }
  dst[idx] = (__bf16)val;
}

// NCHW-flat (c*64+y*8+x) <-> NHWC-flat ((y*8+x)*128+c) remap for the 8192 dim.
__device__ __forceinline__ int remap8192(int j) { return (j & 127) * 64 + (j >> 7); }

__global__ void k_prep_fcv(__bf16* dst, const float* fcv) {    // [32 n][8192 k]
  int idx = blockIdx.x * 256 + threadIdx.x;
  if (idx >= 32 * 8192) return;
  int n = idx >> 13, k = idx & 8191;
  dst[idx] = (__bf16)((n < 10) ? fcv[(size_t)n * 8192 + remap8192(k)] : 0.f);
}
__global__ void k_prep_fcw(__bf16* dst, const float* fcw) {    // [8192 n][64 k]
  int idx = blockIdx.x * 256 + threadIdx.x;
  if (idx >= 8192 * 64) return;
  int n = idx >> 6, k = idx & 63;
  dst[idx] = (__bf16)((k < 10) ? fcw[(size_t)remap8192(n) * 10 + k] : 0.f);
}
__global__ void k_prep_fcvadj(__bf16* dst, const float* fcv) { // [8192 n][64 k]
  int idx = blockIdx.x * 256 + threadIdx.x;
  if (idx >= 8192 * 64) return;
  int n = idx >> 6, k = idx & 63;
  dst[idx] = (__bf16)((k < 10) ? fcv[(size_t)k * 8192 + remap8192(n)] : 0.f);
}
__global__ void k_prep_fcwb(float* dst, const float* fcwb) {
  int n = blockIdx.x * 256 + threadIdx.x;
  if (n < 8192) dst[n] = fcwb[remap8192(n)];
}

__global__ void k_sumsq_bf(const __bf16* p, size_t n, float coef, float* out) {
  __shared__ float red[256];
  size_t base = (size_t)blockIdx.x * blockDim.x * 4 + threadIdx.x;
  float s = 0.f;
  for (int j = 0; j < 4; ++j) {
    size_t ii = base + (size_t)j * blockDim.x;
    if (ii < n) { float v = (float)p[ii]; s += v * v; }
  }
  red[threadIdx.x] = s; __syncthreads();
  for (int st = 128; st > 0; st >>= 1) {
    if ((int)threadIdx.x < st) red[threadIdx.x] += red[threadIdx.x + st];
    __syncthreads();
  }
  if (threadIdx.x == 0) atomicAdd(out, coef * red[0]);
}
__global__ void k_sumsq_f(const float* p, size_t n, float coef, float* out) {
  __shared__ float red[256];
  size_t base = (size_t)blockIdx.x * blockDim.x * 4 + threadIdx.x;
  float s = 0.f;
  for (int j = 0; j < 4; ++j) {
    size_t ii = base + (size_t)j * blockDim.x;
    if (ii < n) { float v = p[ii]; s += v * v; }
  }
  red[threadIdx.x] = s; __syncthreads();
  for (int st = 128; st > 0; st >>= 1) {
    if ((int)threadIdx.x < st) red[threadIdx.x] += red[threadIdx.x + st];
    __syncthreads();
  }
  if (threadIdx.x == 0) atomicAdd(out, coef * red[0]);
}

// ---------------------------------------------------------------------------
// host orchestration
// ---------------------------------------------------------------------------
static inline unsigned cdiv(size_t a, size_t b) { return (unsigned)((a + b - 1) / b); }

static ConvDesc convA(const void* src, int Cin, int lcp,
                      int H, int W, int low, int lohw, int s, int ldil, int pl) {
  ConvDesc d; d.src = src; d.Cin = Cin; d.lcp = lcp; d.H = H; d.W = W;
  d.low = low; d.lohw = lohw; d.s = s; d.ldil = ldil; d.pl = pl; return d;
}
static ConvDesc denseD(const void* src) { ConvDesc d{}; d.src = src; return d; }

static void gemm(hipStream_t st, int mode, const ConvDesc& ad, const __bf16* Bw,
                 int M, int Kpad, int Npad, const float* bias, const float* subf,
                 int N, int ldc, float* oF, __bf16* oB) {
  dim3 g(cdiv(M, MB), Npad / NB);
  switch (mode) {
    case 0: gemm_wmma<0><<<g,256,0,st>>>(ad, Bw, M, Kpad, bias, subf, N, ldc, oF, oB); break;
    case 1: gemm_wmma<1><<<g,256,0,st>>>(ad, Bw, M, Kpad, bias, subf, N, ldc, oF, oB); break;
    case 2: gemm_wmma<2><<<g,256,0,st>>>(ad, Bw, M, Kpad, bias, subf, N, ldc, oF, oB); break;
    case 3: gemm_wmma<3><<<g,256,0,st>>>(ad, Bw, M, Kpad, bias, subf, N, ldc, oF, oB); break;
    case 4: gemm_wmma<4><<<g,256,0,st>>>(ad, Bw, M, Kpad, bias, subf, N, ldc, oF, oB); break;
    default: gemm_wmma<5><<<g,256,0,st>>>(ad, Bw, M, Kpad, bias, subf, N, ldc, oF, oB); break;
  }
}

extern "C" void kernel_launch(void* const* d_in, const int* in_sizes, int n_in,
                              void* d_out, int out_size, void* d_ws, size_t ws_size,
                              hipStream_t stream)
{
  (void)in_sizes; (void)n_in; (void)out_size; (void)ws_size;
  const float* image = (const float*)d_in[0];
  const float* label = (const float*)d_in[1];
  const float* c1v_w = (const float*)d_in[2];  const float* c1v_b = (const float*)d_in[3];
  const float* c2v_w = (const float*)d_in[4];  const float* c2v_b = (const float*)d_in[5];
  const float* c3v_w = (const float*)d_in[6];  const float* c3v_b = (const float*)d_in[7];
  const float* fcv_w = (const float*)d_in[8];  const float* fcv_b = (const float*)d_in[9];
  const float* fcw_w = (const float*)d_in[10]; const float* fcw_b = (const float*)d_in[11];
  const float* c3w_w = (const float*)d_in[12]; const float* c3w_b = (const float*)d_in[13];
  const float* c2w_w = (const float*)d_in[14]; const float* c2w_b = (const float*)d_in[15];
  const float* c1w_w = (const float*)d_in[16]; const float* c1w_b = (const float*)d_in[17];

  const int M1 = 256 * 32 * 32, M2 = 256 * 16 * 16, M3 = 256 * 8 * 8;

  char* ws = (char*)d_ws;
  size_t off = 0;
  auto carve = [&](size_t bytes) -> void* {
    off = (off + 255) & ~(size_t)255;
    void* p = ws + off; off += bytes; return p;
  };
  float*  img_nhwc = (float*)carve((size_t)M1 * 4 * 4);   // 4-channel padded, ch3 = 0
  float*  x1 = (float*)carve((size_t)M1 * 32 * 4);
  float*  v1 = (float*)carve((size_t)M1 * 32 * 4);
  float*  x2 = (float*)carve((size_t)M2 * 64 * 4);
  float*  v2 = (float*)carve((size_t)M2 * 64 * 4);
  float*  x3 = (float*)carve((size_t)M3 * 128 * 4);
  float*  v3 = (float*)carve((size_t)M3 * 128 * 4);
  float*  adjd = (float*)carve((size_t)M1 * 32 * 4);
  float*  adjg = (float*)carve((size_t)M1 * 32 * 4);
  float*  e4f  = (float*)carve(256 * 10 * 4);
  float*  fcwb = (float*)carve(8192 * 4);
  __bf16* e1 = (__bf16*)carve((size_t)M1 * 32 * 2);
  __bf16* f1 = (__bf16*)carve((size_t)M1 * 32 * 2);
  __bf16* e2 = (__bf16*)carve((size_t)M2 * 64 * 2);
  __bf16* f2 = (__bf16*)carve((size_t)M2 * 64 * 2);
  __bf16* e3 = (__bf16*)carve((size_t)M3 * 128 * 2);
  __bf16* f3 = (__bf16*)carve((size_t)M3 * 128 * 2);
  __bf16* f0 = (__bf16*)carve((size_t)M1 * 4 * 2);        // 4-channel padded, ch3 = 0
  __bf16* packE4  = (__bf16*)carve(256 * 64 * 2);
  __bf16* packLab = (__bf16*)carve(256 * 64 * 2);
  __bf16* Wc1v  = (__bf16*)carve((size_t)32 * 64 * 2);
  __bf16* Wc2v  = (__bf16*)carve((size_t)64 * 320 * 2);
  __bf16* Wc3v  = (__bf16*)carve((size_t)128 * 576 * 2);
  __bf16* Wc2vT = (__bf16*)carve((size_t)32 * 576 * 2);
  __bf16* Wc3vT = (__bf16*)carve((size_t)64 * 1152 * 2);
  __bf16* Wc3w  = (__bf16*)carve((size_t)64 * 1152 * 2);
  __bf16* Wc2w  = (__bf16*)carve((size_t)32 * 576 * 2);
  __bf16* Wc1w  = (__bf16*)carve((size_t)32 * 320 * 2);
  __bf16* Wc3wC = (__bf16*)carve((size_t)128 * 576 * 2);
  __bf16* Wc2wC = (__bf16*)carve((size_t)64 * 320 * 2);
  __bf16* Wc1wC = (__bf16*)carve((size_t)32 * 64 * 2);
  __bf16* Wfcv  = (__bf16*)carve((size_t)32 * 8192 * 2);
  __bf16* Wfcw  = (__bf16*)carve((size_t)8192 * 64 * 2);
  __bf16* WfcvA = (__bf16*)carve((size_t)8192 * 64 * 2);

  // ---- one-time prep ----
  {
    size_t nz = (size_t)(M1 * 32) * 2 + (size_t)(M2 * 64) * 2 + (size_t)(M3 * 128) * 2;
    k_zero<<<cdiv(nz, 256), 256, 0, stream>>>(x1, nz);  // x1..v3 contiguous
    size_t ni4 = (size_t)M1 * 4;
    k_zero  <<<cdiv(ni4, 256), 256, 0, stream>>>(img_nhwc, ni4);
    k_zerobf<<<cdiv(ni4, 256), 256, 0, stream>>>(f0, ni4);
    size_t ni = (size_t)M1 * 3;
    k_nchw2nhwc<<<cdiv(ni, 256), 256, 0, stream>>>(img_nhwc, image, 256, 3, 32, 32, 4);
    auto prep = [&](__bf16* dst, const float* w, int Cin, int lcp, int O,
                    int Npad, int Kpad, int mode) {
      k_prep_w<<<cdiv((size_t)Npad * Kpad, 256), 256, 0, stream>>>(dst, w, Cin, lcp, O, Npad, Kpad, mode);
    };
    prep(Wc1v,  c1v_w, 3,   2, 32,  32,  64,   0);
    prep(Wc2v,  c2v_w, 32,  5, 64,  64,  320,  0);
    prep(Wc3v,  c3v_w, 64,  6, 128, 128, 576,  0);
    prep(Wc2vT, c2v_w, 64,  6, 32,  32,  576,  1);
    prep(Wc3vT, c3v_w, 128, 7, 64,  64,  1152, 1);
    prep(Wc3w,  c3w_w, 128, 7, 64,  64,  1152, 1);
    prep(Wc2w,  c2w_w, 64,  6, 32,  32,  576,  1);
    prep(Wc1w,  c1w_w, 32,  5, 3,   32,  320,  1);
    prep(Wc3wC, c3w_w, 64,  6, 128, 128, 576,  0);
    prep(Wc2wC, c2w_w, 32,  5, 64,  64,  320,  0);
    prep(Wc1wC, c1w_w, 3,   2, 32,  32,  64,   0);
    k_prep_fcv   <<<cdiv(32 * 8192, 256), 256, 0, stream>>>(Wfcv,  fcv_w);
    k_prep_fcw   <<<cdiv(8192 * 64, 256), 256, 0, stream>>>(Wfcw,  fcw_w);
    k_prep_fcvadj<<<cdiv(8192 * 64, 256), 256, 0, stream>>>(WfcvA, fcv_w);
    k_prep_fcwb  <<<cdiv(8192, 256), 256, 0, stream>>>(fcwb, fcw_b);
    k_pack<<<cdiv(256 * 64, 256), 256, 0, stream>>>(packLab, label, 256, 10, 64, 1);
  }

  // ---- per-evaluation error pass (8 GEMMs) ----
  auto do_errors = [&]() {
    gemm(stream, 4, convA(img_nhwc, 4, 2, 32, 32, 5, 10, 1, 0, 1),
         Wc1v, M1, 64, 32, c1v_b, x1, 32, 32, nullptr, e1);
    gemm(stream, 2, convA(x1, 32, 5, 32, 32, 4, 8, 2, 0, 1),
         Wc2v, M2, 320, 64, c2v_b, x2, 64, 64, nullptr, e2);
    gemm(stream, 2, convA(x2, 64, 6, 16, 16, 3, 6, 2, 0, 1),
         Wc3v, M3, 576, 128, c3v_b, x3, 128, 128, nullptr, e3);
    gemm(stream, 1, denseD(x3), Wfcv, 256, 8192, 32, fcv_b, label, 10, 10, e4f, nullptr);
    gemm(stream, 0, denseD(packLab), Wfcw, 256, 64, 8192, fcwb, x3, 8192, 8192, nullptr, f3);
    gemm(stream, 2, convA(x3, 128, 7, 8, 8, 4, 8, 1, 1, 1),
         Wc3w, M2, 1152, 64, c3w_b, x2, 64, 64, nullptr, f2);
    gemm(stream, 2, convA(x2, 64, 6, 16, 16, 5, 10, 1, 1, 1),
         Wc2w, M1, 576, 32, c2w_b, x1, 32, 32, nullptr, f1);
    gemm(stream, 2, convA(x1, 32, 5, 32, 32, 5, 10, 1, 0, 1),
         Wc1w, M1, 320, 32, c1w_b, img_nhwc, 3, 4, nullptr, f0);
  };

  // ---- 20 inference steps ----
  for (int step = 0; step < NSTEPS; ++step) {
    do_errors();
    // x1: adjd = convT(e2, c2v_w, s2 p1 op1); adjg = conv(f0, c1w_w, s1 p1)
    gemm(stream, 3, convA(e2, 64, 6, 16, 16, 5, 10, 1, 1, 1),
         Wc2vT, M1, 576, 32, nullptr, nullptr, 32, 32, adjd, nullptr);
    gemm(stream, 5, convA(f0, 4, 2, 32, 32, 5, 10, 1, 0, 1),
         Wc1wC, M1, 64, 32, nullptr, nullptr, 32, 32, adjg, nullptr);
    {
      size_t n = (size_t)M1 * 32;
      k_update<<<cdiv(n, 256), 256, 0, stream>>>(x1, v1, e1, f1, adjd, adjg, n);
    }
    // x2: adjd = convT(e3, c3v_w); adjg = conv(f1, c2w_w, s2 p1)
    gemm(stream, 3, convA(e3, 128, 7, 8, 8, 4, 8, 1, 1, 1),
         Wc3vT, M2, 1152, 64, nullptr, nullptr, 64, 64, adjd, nullptr);
    gemm(stream, 3, convA(f1, 32, 5, 32, 32, 4, 8, 2, 0, 1),
         Wc2wC, M2, 320, 64, nullptr, nullptr, 64, 64, adjg, nullptr);
    {
      size_t n = (size_t)M2 * 64;
      k_update<<<cdiv(n, 256), 256, 0, stream>>>(x2, v2, e2, f2, adjd, adjg, n);
    }
    // x3: adjd = e4 @ fcv_w; adjg = conv(f2, c3w_w, s2 p1)
    k_pack<<<cdiv(256 * 64, 256), 256, 0, stream>>>(packE4, e4f, 256, 10, 64, 0);
    gemm(stream, 0, denseD(packE4), WfcvA, 256, 64, 8192, nullptr, nullptr, 8192, 8192, adjd, nullptr);
    gemm(stream, 3, convA(f2, 64, 6, 16, 16, 3, 6, 2, 0, 1),
         Wc3wC, M3, 576, 128, nullptr, nullptr, 128, 128, adjg, nullptr);
    {
      size_t n = (size_t)M3 * 128;
      k_update<<<cdiv(n, 256), 256, 0, stream>>>(x3, v3, e3, f3, adjd, adjg, n);
    }
  }

  // ---- final energy ----
  do_errors();
  float* out = (float*)d_out;
  k_zero<<<1, 1, 0, stream>>>(out, 1);
  auto ssq_bf = [&](const __bf16* p, size_t n, float c) {
    k_sumsq_bf<<<cdiv(n, 1024), 256, 0, stream>>>(p, n, c, out);
  };
  ssq_bf(e1, (size_t)M1 * 32, 0.5f * AD);
  ssq_bf(e2, (size_t)M2 * 64, 0.5f * AD);
  ssq_bf(e3, (size_t)M3 * 128, 0.5f * AD);
  k_sumsq_f<<<cdiv((size_t)256 * 10, 1024), 256, 0, stream>>>(e4f, 256 * 10, 0.5f * AD, out);
  ssq_bf(f3, (size_t)M3 * 128, 0.5f * AG);
  ssq_bf(f2, (size_t)M2 * 64, 0.5f * AG);
  ssq_bf(f1, (size_t)M1 * 32, 0.5f * AG);
  ssq_bf(f0, (size_t)M1 * 4, 0.5f * AG);   // channel 3 is zero
}
